// MidigenMamba_34883724378129
// MI455X (gfx1250) — matrix operand order repaired and checked by reading, not compiled, not run on hardware
//
#include <hip/hip_runtime.h>
#include <math.h>

#define DIM 768
#define DEPTH 12
#define D_STATE 64
#define D_CONV 4
#define HEADDIM 64
#define VOCAB 2048
#define BB 4
#define LL 2048
#define D_INNER 1536
#define NHEADS 24
#define CONV_DIM 1664            // D_INNER + 2*D_STATE
#define D_IN_PROJ 3224           // 2*D_INNER + 2*D_STATE + NHEADS
#define ROWS (BB * LL)           // 8192
#define EPSL 1e-5f
#define SAPAD 40                 // LDS row stride in halves (80B, 16B-aligned rows)

#if __has_builtin(__builtin_amdgcn_global_load_async_to_lds_b128)
#define USE_ASYNC_LDS 1
#else
#define USE_ASYNC_LDS 0
#endif

typedef __attribute__((ext_vector_type(16))) __bf16 v16bf;
typedef __attribute__((ext_vector_type(8)))  float  v8f;
typedef int v4i_ __attribute__((vector_size(4 * sizeof(int))));

union FragBF { unsigned int u[8]; v16bf v; };

__device__ __forceinline__ unsigned short f2bf(float f) {
  unsigned int u = __float_as_uint(f);
  unsigned int r = u + 0x7FFFu + ((u >> 16) & 1u);   // round-to-nearest-even
  return (unsigned short)(r >> 16);
}

__device__ __forceinline__ float sigm(float x) { return 1.0f / (1.0f + __expf(-x)); }

__device__ __forceinline__ void wait_async_le3() {
#if __has_builtin(__builtin_amdgcn_s_wait_asynccnt)
  __builtin_amdgcn_s_wait_asynccnt(3);
#else
  asm volatile("s_wait_asynccnt 0x3" ::: "memory");
#endif
}
__device__ __forceinline__ void wait_async_le0() {
#if __has_builtin(__builtin_amdgcn_s_wait_asynccnt)
  __builtin_amdgcn_s_wait_asynccnt(0);
#else
  asm volatile("s_wait_asynccnt 0x0" ::: "memory");
#endif
}

// ---------------------------------------------------------------- embed
__global__ void k_embed(const float* __restrict__ tok, const float* __restrict__ pos,
                        const int* __restrict__ ids, float* __restrict__ x) {
  int i = blockIdx.x * blockDim.x + threadIdx.x;
  if (i >= ROWS * DIM) return;
  int d = i % DIM;
  int t = i / DIM;
  int l = t % LL;
  x[i] = tok[(size_t)ids[t] * DIM + d] + pos[(size_t)l * DIM + d];
}

// ---------------------------------------------------------------- f32 -> bf16
__global__ void k_f32_to_bf16(const float* __restrict__ in, unsigned short* __restrict__ out,
                              size_t n) {
  size_t i = (size_t)blockIdx.x * blockDim.x + threadIdx.x;
  if (i < n) out[i] = f2bf(in[i]);
}

// ---------------------------------------------------------------- layernorm -> bf16
__global__ __launch_bounds__(256) void k_layernorm_bf16(
    const float* __restrict__ x, const float* __restrict__ w, const float* __restrict__ b,
    unsigned short* __restrict__ out, int D) {
  __shared__ float red[8];
  int row = blockIdx.x, tid = threadIdx.x;
  const float* xr = x + (size_t)row * D;
  float s = 0.f;
  for (int i = tid; i < D; i += 256) s += xr[i];
  for (int o = 16; o; o >>= 1) s += __shfl_xor(s, o, 32);
  if ((tid & 31) == 0) red[tid >> 5] = s;
  __syncthreads();
  if (tid == 0) { float t = 0.f; for (int i = 0; i < 8; ++i) t += red[i]; red[0] = t; }
  __syncthreads();
  float mean = red[0] / D;
  __syncthreads();
  float v = 0.f;
  for (int i = tid; i < D; i += 256) { float d = xr[i] - mean; v += d * d; }
  for (int o = 16; o; o >>= 1) v += __shfl_xor(v, o, 32);
  if ((tid & 31) == 0) red[tid >> 5] = v;
  __syncthreads();
  if (tid == 0) { float t = 0.f; for (int i = 0; i < 8; ++i) t += red[i]; red[0] = t; }
  __syncthreads();
  float rstd = rsqrtf(red[0] / D + EPSL);
  unsigned short* orow = out + (size_t)row * D;
  for (int i = tid; i < D; i += 256)
    orow[i] = f2bf((xr[i] - mean) * rstd * w[i] + b[i]);
}

// ---------------------------------------------------------------- WMMA GEMM
// C[M,N] = A[M,K](bf16) @ Bw[N,K]^T(bf16)  (+ addsrc), f32 accumulate.
// Block tile 128x64, 8 waves (4x2), each wave: 32x32 via 2x2 v_wmma_f32_16x16x32_bf16.
// Tiles staged via double-buffered global_load_async_to_lds_b128 (ASYNCcnt pipeline)
// when available; else fully unrolled uint4 register staging.
__global__ __launch_bounds__(256) void k_gemm_bf16(
    const unsigned short* __restrict__ A,   // M x K row-major bf16
    const unsigned short* __restrict__ Bw,  // N x K row-major bf16 (weight)
    const float* __restrict__ addsrc,       // nullable, M x N
    float* __restrict__ C, int M, int N, int K) {
  (void)M;
  __shared__ __align__(16) unsigned short sA[2][128][SAPAD];
  __shared__ __align__(16) unsigned short sB[2][64][SAPAD];
#if USE_ASYNC_LDS
  __shared__ __align__(16) unsigned short sDummy[8];
#endif
  const int tid  = threadIdx.x;
  const int lane = tid & 31;
  const int wave = tid >> 5;
  const int wm = wave >> 1;   // 0..3
  const int wn = wave & 1;    // 0..1
  const int gm0 = blockIdx.y * 128;
  const int gn0 = blockIdx.x * 64;
  const int hf = lane >> 4;
  const int lr = lane & 15;

  // per-thread 16B staging chunks: A tile = 512 chunks (2/thread), B tile = 256 (1/thread)
  const int am0 = tid >> 2, am1 = am0 + 64, aj = tid & 3;
  const int bn  = tid >> 2, bj  = tid & 3;

  v8f acc[2][2] = {};
  const int nk = K >> 5;

  auto compute = [&](int bf) {
    FragBF a[2], b[2];
#pragma unroll
    for (int i = 0; i < 2; ++i) {
      int m = wm * 32 + i * 16 + lr;
#pragma unroll
      for (int v = 0; v < 8; ++v) {
        int k = 16 * (v >> 2) + 2 * (v & 3) + 8 * hf;   // CDNA5 bf16 A 16x32 layout
        a[i].u[v] = *(const unsigned int*)&sA[bf][m][k];
      }
    }
#pragma unroll
    for (int j = 0; j < 2; ++j) {
      int n = wn * 32 + j * 16 + lr;
#pragma unroll
      for (int v = 0; v < 8; ++v) {
        int k = 16 * (v >> 2) + 2 * (v & 3) + 8 * hf;   // CDNA5 bf16 B 32x16 layout
        b[j].u[v] = *(const unsigned int*)&sB[bf][n][k];
      }
    }
#pragma unroll
    for (int i = 0; i < 2; ++i)
#pragma unroll
      for (int j = 0; j < 2; ++j)
        acc[i][j] = __builtin_amdgcn_wmma_f32_16x16x32_bf16(
            false, a[i].v, false, b[j].v, (short)0, acc[i][j], false, false);
  };

#if USE_ASYNC_LDS
  // ragged N: pre-zero OOB sB rows once; async loads never touch them
  if (gn0 + 64 > N) {
    for (int idx = tid; idx < 512; idx += 256) {
      int bf = idx >> 8, rem = idx & 255, n = rem >> 2, j = rem & 3;
      if (gn0 + n >= N) *(uint4*)&sB[bf][n][j * 8] = make_uint4(0u, 0u, 0u, 0u);
    }
  }
  auto issue_tile = [&](int kt, int bf) {
    __builtin_amdgcn_global_load_async_to_lds_b128(
        (__attribute__((address_space(1))) v4i_*)(A + (size_t)(gm0 + am0) * K + kt + aj * 8),
        (__attribute__((address_space(3))) v4i_*)&sA[bf][am0][aj * 8], 0, 0);
    __builtin_amdgcn_global_load_async_to_lds_b128(
        (__attribute__((address_space(1))) v4i_*)(A + (size_t)(gm0 + am1) * K + kt + aj * 8),
        (__attribute__((address_space(3))) v4i_*)&sA[bf][am1][aj * 8], 0, 0);
    // uniform op count per wave: OOB lanes fetch a safe row into a dummy slot
    bool ok = (gn0 + bn) < N;
    const unsigned short* gsrc =
        Bw + (size_t)(ok ? (gn0 + bn) : (N - 1)) * K + kt + bj * 8;
    unsigned short* ldst = ok ? &sB[bf][bn][bj * 8] : &sDummy[0];
    __builtin_amdgcn_global_load_async_to_lds_b128(
        (__attribute__((address_space(1))) v4i_*)gsrc,
        (__attribute__((address_space(3))) v4i_*)ldst, 0, 0);
  };

  issue_tile(0, 0);
  for (int t = 0; t < nk; ++t) {
    const int bf = t & 1;
    if (t + 1 < nk) {
      issue_tile((t + 1) << 5, bf ^ 1);   // prefetch next tile into other buffer
      wait_async_le3();                    // my 3 ops for tile t are done
    } else {
      wait_async_le0();
    }
    __syncthreads();                       // everyone's tile t is in LDS
    compute(bf);
    __syncthreads();                       // reads done before buffer reuse
  }
#else
  for (int kt = 0; kt < K; kt += 32) {
    uint4 ra0 = *(const uint4*)(A + (size_t)(gm0 + am0) * K + kt + aj * 8);
    uint4 ra1 = *(const uint4*)(A + (size_t)(gm0 + am1) * K + kt + aj * 8);
    uint4 rb = make_uint4(0u, 0u, 0u, 0u);
    if (gn0 + bn < N)
      rb = *(const uint4*)(Bw + (size_t)(gn0 + bn) * K + kt + bj * 8);
    __syncthreads();                       // previous iteration's reads done
    *(uint4*)&sA[0][am0][aj * 8] = ra0;
    *(uint4*)&sA[0][am1][aj * 8] = ra1;
    *(uint4*)&sB[0][bn][bj * 8] = rb;
    __syncthreads();
    compute(0);
  }
#endif

#pragma unroll
  for (int i = 0; i < 2; ++i) {
#pragma unroll
    for (int j = 0; j < 2; ++j) {
      int n = gn0 + wn * 32 + j * 16 + lr;
      if (n < N) {
#pragma unroll
        for (int v = 0; v < 8; ++v) {
          int m = gm0 + wm * 32 + i * 16 + v + 8 * hf;  // CDNA5 f32 C/D layout
          size_t off = (size_t)m * N + n;
          float val = acc[i][j][v];
          if (addsrc) val += addsrc[off];
          C[off] = val;
        }
      }
    }
  }
}

// ---------------------------------------------------------------- depthwise conv + SiLU
__global__ void k_conv_silu(const float* __restrict__ zx, const float* __restrict__ cw,
                            const float* __restrict__ cb, float* __restrict__ xact) {
  int i = blockIdx.x * blockDim.x + threadIdx.x;
  if (i >= ROWS * CONV_DIM) return;
  int c = i % CONV_DIM;
  int r = i / CONV_DIM;
  int l = r % LL;
  int b = r / LL;
  float acc = cb[c];
#pragma unroll
  for (int j = 0; j < D_CONV; ++j) {
    int lt = l - (D_CONV - 1) + j;
    if (lt >= 0)
      acc += zx[(size_t)(b * LL + lt) * D_IN_PROJ + D_INNER + c] * cw[c * D_CONV + j];
  }
  xact[i] = acc * sigm(acc);
}

// ---------------------------------------------------------------- dt = softplus, dA = exp(dt*A)
__global__ void k_dt(const float* __restrict__ zx, const float* __restrict__ dtb,
                     const float* __restrict__ alog, float* __restrict__ dtbuf,
                     float* __restrict__ dAbuf) {
  int i = blockIdx.x * blockDim.x + threadIdx.x;
  if (i >= ROWS * NHEADS) return;
  int h = i % NHEADS;
  int r = i / NHEADS;
  float raw = zx[(size_t)r * D_IN_PROJ + (D_INNER + CONV_DIM) + h] + dtb[h];
  float dt = (raw > 20.f) ? raw : log1pf(__expf(raw));
  float A = -__expf(alog[h]);
  dtbuf[i] = dt;
  dAbuf[i] = __expf(dt * A);
}

// ---------------------------------------------------------------- sequential SSM scan
__global__ __launch_bounds__(256) void k_scan(
    const float* __restrict__ xact, const float* __restrict__ dtbuf,
    const float* __restrict__ dAbuf, const float* __restrict__ Dp,
    float* __restrict__ y) {
  __shared__ float sX[HEADDIM], sBv[D_STATE], sCv[D_STATE], sS[2];
  int b = blockIdx.x / NHEADS, h = blockIdx.x % NHEADS;
  int tid = threadIdx.x;
  int p = tid >> 2, q = tid & 3, n0 = q << 4;
  float dh = Dp[h];
  float hs[16];
#pragma unroll
  for (int j = 0; j < 16; ++j) hs[j] = 0.f;
  for (int l = 0; l < LL; ++l) {
    size_t rb = (size_t)(b * LL + l) * CONV_DIM;
    if (tid < 64)        sX[tid] = xact[rb + h * HEADDIM + tid];
    else if (tid < 128)  sBv[tid - 64] = xact[rb + D_INNER + (tid - 64)];
    else if (tid < 192)  sCv[tid - 128] = xact[rb + D_INNER + D_STATE + (tid - 128)];
    else if (tid == 192) {
      int gi = (b * LL + l) * NHEADS + h;
      sS[0] = dAbuf[gi];
      sS[1] = dtbuf[gi];
    }
    __syncthreads();
    float dAt = sS[0];
    float dtx = sS[1] * sX[p];
    float part = 0.f;
#pragma unroll
    for (int j = 0; j < 16; ++j) {
      hs[j] = hs[j] * dAt + dtx * sBv[n0 + j];
      part += hs[j] * sCv[n0 + j];
    }
    part += __shfl_xor(part, 1, 32);
    part += __shfl_xor(part, 2, 32);
    if (q == 0)
      y[(size_t)(b * LL + l) * D_INNER + h * HEADDIM + p] = part + dh * sX[p];
    __syncthreads();
  }
}

// ---------------------------------------------------------------- gated RMSNorm -> bf16
__global__ __launch_bounds__(256) void k_gated_rms(
    const float* __restrict__ y, const float* __restrict__ zx,
    const float* __restrict__ nw, unsigned short* __restrict__ out) {
  __shared__ float red[8];
  int row = blockIdx.x, tid = threadIdx.x;
  const float* yr = y + (size_t)row * D_INNER;
  const float* zr = zx + (size_t)row * D_IN_PROJ;   // z = first D_INNER cols
  float s = 0.f;
  for (int i = tid; i < D_INNER; i += 256) {
    float z = zr[i];
    float g = yr[i] * (z * sigm(z));
    s += g * g;
  }
  for (int o = 16; o; o >>= 1) s += __shfl_xor(s, o, 32);
  if ((tid & 31) == 0) red[tid >> 5] = s;
  __syncthreads();
  if (tid == 0) { float t = 0.f; for (int i = 0; i < 8; ++i) t += red[i]; red[0] = t; }
  __syncthreads();
  float scale = rsqrtf(red[0] / D_INNER + EPSL);
  unsigned short* orow = out + (size_t)row * D_INNER;
  for (int i = tid; i < D_INNER; i += 256) {
    float z = zr[i];
    float g = yr[i] * (z * sigm(z));
    orow[i] = f2bf(g * scale * nw[i]);
  }
}

// ---------------------------------------------------------------- cross-entropy
__global__ __launch_bounds__(256) void k_loss_rows(
    const float* __restrict__ logits, const int* __restrict__ labels,
    float* __restrict__ rowloss) {
  __shared__ float red[8];
  int r = blockIdx.x;                 // 0 .. BB*(LL-1)-1
  int b = r / (LL - 1), l = r % (LL - 1);
  const float* lg = logits + (size_t)(b * LL + l) * VOCAB;
  int tid = threadIdx.x;
  float m = -1e30f;
  for (int i = tid; i < VOCAB; i += 256) m = fmaxf(m, lg[i]);
  for (int o = 16; o; o >>= 1) m = fmaxf(m, __shfl_xor(m, o, 32));
  if ((tid & 31) == 0) red[tid >> 5] = m;
  __syncthreads();
  if (tid == 0) { float t = red[0]; for (int i = 1; i < 8; ++i) t = fmaxf(t, red[i]); red[0] = t; }
  __syncthreads();
  float mx = red[0];
  __syncthreads();
  float s = 0.f;
  for (int i = tid; i < VOCAB; i += 256) s += __expf(lg[i] - mx);
  for (int o = 16; o; o >>= 1) s += __shfl_xor(s, o, 32);
  if ((tid & 31) == 0) red[tid >> 5] = s;
  __syncthreads();
  if (tid == 0) {
    float t = 0.f;
    for (int i = 0; i < 8; ++i) t += red[i];
    int tgt = labels[b * LL + l + 1];
    rowloss[r] = logf(t) + mx - lg[tgt];
  }
}

__global__ __launch_bounds__(256) void k_loss_reduce(const float* __restrict__ rowloss,
                                                     float* __restrict__ out, int n) {
  __shared__ float red[8];
  int tid = threadIdx.x;
  float s = 0.f;
  for (int i = tid; i < n; i += 256) s += rowloss[i];
  for (int o = 16; o; o >>= 1) s += __shfl_xor(s, o, 32);
  if ((tid & 31) == 0) red[tid >> 5] = s;
  __syncthreads();
  if (tid == 0) {
    float t = 0.f;
    for (int i = 0; i < 8; ++i) t += red[i];
    out[0] = t / n;
  }
}

// ================================================================ host
extern "C" void kernel_launch(void* const* d_in, const int* in_sizes, int n_in,
                              void* d_out, int out_size, void* d_ws, size_t ws_size,
                              hipStream_t stream) {
  (void)in_sizes; (void)n_in; (void)out_size; (void)ws_size;
  const float* tok  = (const float*)d_in[0];
  const float* pos  = (const float*)d_in[1];
  const float* ln_w = (const float*)d_in[2];
  const float* ln_b = (const float*)d_in[3];
  const float* in_w = (const float*)d_in[4];
  const float* cw   = (const float*)d_in[5];
  const float* cb   = (const float*)d_in[6];
  const float* dtb  = (const float*)d_in[7];
  const float* alog = (const float*)d_in[8];
  const float* Dp   = (const float*)d_in[9];
  const float* nw   = (const float*)d_in[10];
  const float* ow   = (const float*)d_in[11];
  const float* nfw  = (const float*)d_in[12];
  const float* nfb  = (const float*)d_in[13];
  const float* hw   = (const float*)d_in[14];
  const int* ids    = (const int*)d_in[15];
  const int* labels = (const int*)d_in[16];
  float* out = (float*)d_out;
  float* logits = out + 1;   // (loss, logits) flat

  char* ws = (char*)d_ws;
  size_t off = 0;
  auto alloc = [&](size_t bytes) -> void* {
    void* p = ws + off;
    off += (bytes + 255) & ~(size_t)255;
    return p;
  };
  float* x              = (float*)alloc((size_t)ROWS * DIM * 4);
  unsigned short* lnb   = (unsigned short*)alloc((size_t)ROWS * DIM * 2);
  float* zx             = (float*)alloc((size_t)ROWS * D_IN_PROJ * 4);
  float* xact           = (float*)alloc((size_t)ROWS * CONV_DIM * 4);
  float* dtbuf          = (float*)alloc((size_t)ROWS * NHEADS * 4);
  float* dAbuf          = (float*)alloc((size_t)ROWS * NHEADS * 4);
  float* ybuf           = (float*)alloc((size_t)ROWS * D_INNER * 4);
  unsigned short* ybf   = (unsigned short*)alloc((size_t)ROWS * D_INNER * 2);
  unsigned short* winb  = (unsigned short*)alloc((size_t)DEPTH * D_IN_PROJ * DIM * 2);
  unsigned short* wob   = (unsigned short*)alloc((size_t)DEPTH * DIM * D_INNER * 2);
  unsigned short* whb   = (unsigned short*)alloc((size_t)VOCAB * DIM * 2);
  float* rowloss        = (float*)alloc((size_t)BB * (LL - 1) * 4);

  // weight conversion f32 -> bf16 (deterministic, every call)
  {
    size_t n1 = (size_t)DEPTH * D_IN_PROJ * DIM;
    k_f32_to_bf16<<<(unsigned)((n1 + 255) / 256), 256, 0, stream>>>(in_w, winb, n1);
    size_t n2 = (size_t)DEPTH * DIM * D_INNER;
    k_f32_to_bf16<<<(unsigned)((n2 + 255) / 256), 256, 0, stream>>>(ow, wob, n2);
    size_t n3 = (size_t)VOCAB * DIM;
    k_f32_to_bf16<<<(unsigned)((n3 + 255) / 256), 256, 0, stream>>>(hw, whb, n3);
  }

  k_embed<<<(ROWS * DIM + 255) / 256, 256, 0, stream>>>(tok, pos, ids, x);

  for (int layer = 0; layer < DEPTH; ++layer) {
    k_layernorm_bf16<<<ROWS, 256, 0, stream>>>(x, ln_w + layer * DIM, ln_b + layer * DIM,
                                               lnb, DIM);
    dim3 g1((D_IN_PROJ + 63) / 64, ROWS / 128);
    k_gemm_bf16<<<g1, 256, 0, stream>>>(lnb, winb + (size_t)layer * D_IN_PROJ * DIM,
                                        nullptr, zx, ROWS, D_IN_PROJ, DIM);
    k_conv_silu<<<(unsigned)(((size_t)ROWS * CONV_DIM + 255) / 256), 256, 0, stream>>>(
        zx, cw + (size_t)layer * CONV_DIM * D_CONV, cb + layer * CONV_DIM, xact);
    k_dt<<<(ROWS * NHEADS + 255) / 256, 256, 0, stream>>>(
        zx, dtb + layer * NHEADS, alog + layer * NHEADS, dtbuf, dAbuf);
    k_scan<<<BB * NHEADS, 256, 0, stream>>>(xact, dtbuf, dAbuf, Dp + layer * NHEADS, ybuf);
    k_gated_rms<<<ROWS, 256, 0, stream>>>(ybuf, zx, nw + layer * D_INNER, ybf);
    dim3 g2((DIM + 63) / 64, ROWS / 128);
    k_gemm_bf16<<<g2, 256, 0, stream>>>(ybf, wob + (size_t)layer * DIM * D_INNER,
                                        x, x, ROWS, DIM, D_INNER);   // fused residual
  }

  k_layernorm_bf16<<<ROWS, 256, 0, stream>>>(x, nfw, nfb, lnb, DIM);
  dim3 g3((VOCAB + 63) / 64, ROWS / 128);
  k_gemm_bf16<<<g3, 256, 0, stream>>>(lnb, whb, nullptr, logits, ROWS, VOCAB, DIM);

  int nrows = BB * (LL - 1);
  k_loss_rows<<<nrows, 256, 0, stream>>>(logits, labels, rowloss);
  k_loss_reduce<<<1, 256, 0, stream>>>(rowloss, out, nrows);
}
